// Seq2Seq_76149770158196
// MI455X (gfx1250) — compile-verified, matrix operand order checked by reading, not run on hardware
//
#include <hip/hip_runtime.h>
#include <stdint.h>

// ---------------- model constants ----------------
#define BB   32
#define TT   128
#define EE   512
#define HH   1024
#define AA   1024
#define VV   32000
#define SS   100           // MAX_SEQ
#define G3H  3072          // 3*H

typedef __attribute__((ext_vector_type(16))) __bf16 v16bf;
typedef __attribute__((ext_vector_type(8)))  float  v8f;

union BfPack { v16bf v; uint32_t u[8]; };

__device__ __forceinline__ uint16_t f2bf(float f) {
    uint32_t u = __float_as_uint(f);
    uint32_t r = (u + 0x7FFFu + ((u >> 16) & 1u)) >> 16;   // RNE
    return (uint16_t)r;
}

// ---------------- fp32 -> bf16 convert ----------------
__global__ void k_cvt(const float* __restrict__ s, uint16_t* __restrict__ d, int n) {
    int i = blockIdx.x * blockDim.x + threadIdx.x;
    if (i < n) d[i] = f2bf(s[i]);
}

// ---------------- encoder embedding gather (fp32 -> bf16) ----------------
__global__ void k_embed(const int* __restrict__ inputs, const float* __restrict__ emb,
                        uint16_t* __restrict__ xbf) {
    int i = blockIdx.x * blockDim.x + threadIdx.x;        // over B*T*E
    if (i >= BB * TT * EE) return;
    int tok = i >> 9;                                     // E = 512
    int c   = i & 511;
    xbf[i] = f2bf(emb[(size_t)inputs[tok] * EE + c]);
}

// ---------------- generic bf16 WMMA GEMM: C[M,N] = A[M,K] * W[N,K]^T + bias ----
// Register-blocked along N: one wave owns 1 M-tile x NT N-tiles, loads the A
// fragment once per k-step and issues NT back-to-back v_wmma ops against it.
// A layout per CDNA5 ISA: lane = M%16, lane-half selects K+8, dword v holds a
// consecutive K pair at (v/4)*16 + hi*8 + (v%4)*2; B mirrors A with lane = N%16.
// D layout: VGPR i -> row M = i + 8*hi, col N = lane%16.
template <int NT>
__global__ __launch_bounds__(128)
void k_gemm(const uint16_t* __restrict__ A, int lda,
            const uint16_t* __restrict__ W, int ldw,
            const float* __restrict__ bias,
            float* __restrict__ Cf, uint16_t* __restrict__ Cbf, int ldc,
            int M, int N, int K, int act) {
    int wavesPerBlock = blockDim.x >> 5;
    int wave   = blockIdx.x * wavesPerBlock + (threadIdx.x >> 5);
    int tilesN  = N >> 4;
    int groupsN = tilesN / NT;
    int total   = (M >> 4) * groupsN;
    if (wave >= total) return;                 // wave-uniform: EXEC stays all-1s
    int tm = wave / groupsN;
    int tg = wave % groupsN;
    int lane = threadIdx.x & 31;
    int r  = lane & 15;
    int hi = lane >> 4;

    const uint16_t* arow = A + (size_t)(tm * 16 + r) * lda;
    const uint16_t* wrow[NT];
#pragma unroll
    for (int nt = 0; nt < NT; ++nt)
        wrow[nt] = W + (size_t)((tg * NT + nt) * 16 + r) * ldw;

    v8f acc[NT] = {};

    for (int k0 = 0; k0 < K; k0 += 32) {
        BfPack a;
#pragma unroll
        for (int v = 0; v < 8; ++v) {
            int kk = k0 + ((v >> 2) << 4) + (hi << 3) + ((v & 3) << 1);
            a.u[v] = *(const uint32_t*)(arow + kk);
        }
        BfPack b[NT];
#pragma unroll
        for (int nt = 0; nt < NT; ++nt) {
#pragma unroll
            for (int v = 0; v < 8; ++v) {
                int kk = k0 + ((v >> 2) << 4) + (hi << 3) + ((v & 3) << 1);
                b[nt].u[v] = *(const uint32_t*)(wrow[nt] + kk);
            }
            if (k0 + 32 < K) {
                __builtin_prefetch(wrow[nt] + k0 + 64, 0, 1); // global_prefetch_b8
            }
        }
#pragma unroll
        for (int nt = 0; nt < NT; ++nt) {
            acc[nt] = __builtin_amdgcn_wmma_f32_16x16x32_bf16(
                false, a.v, false, b[nt].v, (short)0, acc[nt], false, false);
        }
    }

#pragma unroll
    for (int nt = 0; nt < NT; ++nt) {
        int col = (tg * NT + nt) * 16 + r;
        float bcol = bias ? bias[col] : 0.0f;
#pragma unroll
        for (int i = 0; i < 8; ++i) {
            int row = tm * 16 + (hi << 3) + i;
            float val = acc[nt][i] + bcol;
            if (act == 1) val = tanhf(val);
            size_t off = (size_t)row * ldc + col;
            if (Cf)  Cf[off]  = val;
            if (Cbf) Cbf[off] = f2bf(val);
        }
    }
}

// ---------------- GRU gate math (PyTorch cell) ----------------
// gi rows: stride giStride (encoder packs [B,T,3H]); gh rows: 3H.
__global__ void k_gru(const float* __restrict__ gi, int giStride,
                      const float* __restrict__ gh,
                      float* __restrict__ h, uint16_t* __restrict__ hbf,
                      float* __restrict__ encout, int encStride) {
    int i = blockIdx.x * blockDim.x + threadIdx.x;        // B*H
    if (i >= BB * HH) return;
    int b = i >> 10;                                      // H = 1024
    int j = i & 1023;
    const float* gir = gi + (size_t)b * giStride;
    const float* ghr = gh + (size_t)b * G3H;
    float ir = gir[j], iz = gir[HH + j], in_ = gir[2 * HH + j];
    float hr = ghr[j], hz = ghr[HH + j], hn  = ghr[2 * HH + j];
    float rg = 1.0f / (1.0f + __expf(-(ir + hr)));
    float zg = 1.0f / (1.0f + __expf(-(iz + hz)));
    float ng = tanhf(in_ + rg * hn);
    float hv = (1.0f - zg) * ng + zg * h[i];
    h[i]   = hv;
    hbf[i] = f2bf(hv);
    if (encout) encout[(size_t)b * encStride + j] = hv;
}

// ---------------- decoder input build: x = [dec_emb[ids], attn_vec] (bf16) ---
__global__ void k_build_x(const int* __restrict__ ids, const float* __restrict__ dec_emb,
                          const uint16_t* __restrict__ avbf, uint16_t* __restrict__ xbf) {
    int i = blockIdx.x * blockDim.x + threadIdx.x;        // B*(E+A) = 32*1536
    if (i >= BB * (EE + AA)) return;
    int b = i / (EE + AA);
    int c = i % (EE + AA);
    uint16_t v;
    if (c < EE) v = f2bf(dec_emb[(size_t)ids[b] * EE + c]);
    else        v = avbf[b * AA + (c - EE)];
    xbf[i] = v;
}

// ---------------- attention: score / softmax / context; emit cat=[ctx,h2] bf16
__global__ void k_attention(const float* __restrict__ encOut,   // [B,T,H]
                            const float* __restrict__ h2,       // [B,H]
                            uint16_t* __restrict__ catbf,       // [B,2H] bf16
                            float* __restrict__ w_out) {        // [B,T] slice of d_out
    int b = blockIdx.x;
    int tid = threadIdx.x;                                // blockDim = 128 = T
    __shared__ float h2s[HH];
    __shared__ float sc[TT];
    __shared__ float red[TT];
    for (int j = tid; j < HH; j += TT) h2s[j] = h2[(size_t)b * HH + j];
    __syncthreads();

    const float* eb = encOut + (size_t)b * TT * HH;
    const float* row = eb + (size_t)tid * HH;
    float s = 0.0f;
    for (int j = 0; j < HH; ++j) s += row[j] * h2s[j];
    sc[tid] = s; red[tid] = s;
    __syncthreads();
    for (int off = 64; off > 0; off >>= 1) {
        if (tid < off) red[tid] = fmaxf(red[tid], red[tid + off]);
        __syncthreads();
    }
    float mx = red[0];
    __syncthreads();
    float e = __expf(sc[tid] - mx);
    red[tid] = e;
    __syncthreads();
    for (int off = 64; off > 0; off >>= 1) {
        if (tid < off) red[tid] += red[tid + off];
        __syncthreads();
    }
    float w = e / red[0];
    __syncthreads();
    sc[tid] = w;
    __syncthreads();
    w_out[(size_t)b * TT + tid] = w;

    for (int j = tid; j < HH; j += TT) {
        float c = 0.0f;
        for (int t = 0; t < TT; ++t) c += sc[t] * eb[(size_t)t * HH + j];
        catbf[(size_t)b * 2 * HH + j]      = f2bf(c);
        catbf[(size_t)b * 2 * HH + HH + j] = f2bf(h2s[j]);
    }
}

// ---------------- argmax over V (first-max-index semantics) ----------------
__global__ void k_argmax(const float* __restrict__ preds, int* __restrict__ ids) {
    int b = blockIdx.x;
    int tid = threadIdx.x;                                // 256
    const float* row = preds + (size_t)b * VV;
    float bv = -3.402823466e38f;
    int   bi = 0x7fffffff;
    for (int v = tid; v < VV; v += 256) {
        float x = row[v];
        if (x > bv || (x == bv && v < bi)) { bv = x; bi = v; }
    }
    __shared__ float sv[256];
    __shared__ int   si[256];
    sv[tid] = bv; si[tid] = bi;
    __syncthreads();
    for (int off = 128; off > 0; off >>= 1) {
        if (tid < off) {
            if (sv[tid + off] > sv[tid] ||
                (sv[tid + off] == sv[tid] && si[tid + off] < si[tid])) {
                sv[tid] = sv[tid + off]; si[tid] = si[tid + off];
            }
        }
        __syncthreads();
    }
    if (tid == 0) ids[b] = si[0];
}

// =====================================================================
static inline int cdiv(long long a, long long b) { return (int)((a + b - 1) / b); }

extern "C" void kernel_launch(void* const* d_in, const int* in_sizes, int n_in,
                              void* d_out, int out_size, void* d_ws, size_t ws_size,
                              hipStream_t stream) {
    (void)in_sizes; (void)n_in; (void)out_size; (void)ws_size;

    const int*   inputs  = (const int*)  d_in[0];
    const float* enc_emb = (const float*)d_in[1];
    const float* enc_Wih = (const float*)d_in[2];
    const float* enc_Whh = (const float*)d_in[3];
    const float* enc_bih = (const float*)d_in[4];
    const float* enc_bhh = (const float*)d_in[5];
    const float* dec_emb = (const float*)d_in[6];
    const float* dec_Wih = (const float*)d_in[7];
    const float* dec_Whh = (const float*)d_in[8];
    const float* dec_bih = (const float*)d_in[9];
    const float* dec_bhh = (const float*)d_in[10];
    const float* attn_W  = (const float*)d_in[11];
    const float* attn_b  = (const float*)d_in[12];
    const float* cls_W   = (const float*)d_in[13];
    const float* cls_b   = (const float*)d_in[14];

    float* preds_out = (float*)d_out;                           // [100,B,V]
    float* w_out     = preds_out + (size_t)SS * BB * VV;        // [100,B,T]

    // ---------------- workspace carve-out ----------------
    uint8_t* ws = (uint8_t*)d_ws;
    size_t cur = 0;
    auto carve = [&](size_t bytes) -> uint8_t* {
        cur = (cur + 255) & ~(size_t)255;
        uint8_t* p = ws + cur;
        cur += bytes;
        return p;
    };
    uint16_t* encWih_bf = (uint16_t*)carve((size_t)G3H * EE * 2);
    uint16_t* encWhh_bf = (uint16_t*)carve((size_t)G3H * HH * 2);
    uint16_t* decWih_bf = (uint16_t*)carve((size_t)G3H * (EE + AA) * 2);
    uint16_t* decWhh_bf = (uint16_t*)carve((size_t)G3H * HH * 2);
    uint16_t* attnW_bf  = (uint16_t*)carve((size_t)AA * 2 * HH * 2);
    uint16_t* clsW_bf   = (uint16_t*)carve((size_t)VV * AA * 2);
    uint16_t* x_enc_bf  = (uint16_t*)carve((size_t)BB * TT * EE * 2);
    float*    gi_enc    = (float*)   carve((size_t)BB * TT * G3H * 4);
    float*    gh_buf    = (float*)   carve((size_t)BB * G3H * 4);
    float*    gi_dec    = (float*)   carve((size_t)BB * G3H * 4);
    float*    h_f32     = (float*)   carve((size_t)BB * HH * 4);
    uint16_t* h_bf      = (uint16_t*)carve((size_t)BB * HH * 2);
    float*    enc_out   = (float*)   carve((size_t)BB * TT * HH * 4);
    uint16_t* x_dec_bf  = (uint16_t*)carve((size_t)BB * (EE + AA) * 2);
    uint16_t* av_bf     = (uint16_t*)carve((size_t)BB * AA * 2);
    uint16_t* cat_bf    = (uint16_t*)carve((size_t)BB * 2 * HH * 2);
    int*      ids       = (int*)     carve((size_t)BB * 4);

    // ---------------- zero-init state ----------------
    hipMemsetAsync(h_f32, 0, (size_t)BB * HH * 4, stream);
    hipMemsetAsync(h_bf,  0, (size_t)BB * HH * 2, stream);
    hipMemsetAsync(av_bf, 0, (size_t)BB * AA * 2, stream);
    hipMemsetAsync(ids,   0, (size_t)BB * 4,      stream);

    // ---------------- weight conversion fp32 -> bf16 ----------------
    auto cvt = [&](const float* s, uint16_t* d, long long n) {
        k_cvt<<<cdiv(n, 256), 256, 0, stream>>>(s, d, (int)n);
    };
    cvt(enc_Wih, encWih_bf, (long long)G3H * EE);
    cvt(enc_Whh, encWhh_bf, (long long)G3H * HH);
    cvt(dec_Wih, decWih_bf, (long long)G3H * (EE + AA));
    cvt(dec_Whh, decWhh_bf, (long long)G3H * HH);
    cvt(attn_W,  attnW_bf,  (long long)AA * 2 * HH);
    cvt(cls_W,   clsW_bf,   (long long)VV * AA);

    // ---------------- encoder ----------------
    k_embed<<<cdiv((long long)BB * TT * EE, 256), 256, 0, stream>>>(inputs, enc_emb, x_enc_bf);

    // nt=4: throughput GEMMs (A-fragment reuse); nt=1: latency-bound recurrent GEMMs
    auto gemm = [&](const uint16_t* A, int lda, const uint16_t* W, int ldw,
                    const float* bias, float* Cf, uint16_t* Cbf, int ldc,
                    int M, int N, int K, int act, int nt) {
        if (nt == 4) {
            int waves = (M >> 4) * ((N >> 4) / 4);
            k_gemm<4><<<cdiv(waves, 4), 128, 0, stream>>>(A, lda, W, ldw, bias,
                                                          Cf, Cbf, ldc, M, N, K, act);
        } else {
            int waves = (M >> 4) * (N >> 4);
            k_gemm<1><<<cdiv(waves, 4), 128, 0, stream>>>(A, lda, W, ldw, bias,
                                                          Cf, Cbf, ldc, M, N, K, act);
        }
    };

    // hoisted: GI = emb @ Wih^T + bih over all (b,t)   [4096 x 3072]
    gemm(x_enc_bf, EE, encWih_bf, EE, enc_bih, gi_enc, nullptr, G3H,
         BB * TT, G3H, EE, 0, 4);

    for (int t = 0; t < TT; ++t) {
        gemm(h_bf, HH, encWhh_bf, HH, enc_bhh, gh_buf, nullptr, G3H,
             BB, G3H, HH, 0, 1);
        k_gru<<<cdiv(BB * HH, 256), 256, 0, stream>>>(
            gi_enc + (size_t)t * G3H, TT * G3H, gh_buf,
            h_f32, h_bf, enc_out + (size_t)t * HH, TT * HH);
    }

    // ---------------- greedy decoder, 100 steps ----------------
    for (int s = 0; s < SS; ++s) {
        k_build_x<<<cdiv(BB * (EE + AA), 256), 256, 0, stream>>>(ids, dec_emb, av_bf, x_dec_bf);
        gemm(x_dec_bf, EE + AA, decWih_bf, EE + AA, dec_bih, gi_dec, nullptr, G3H,
             BB, G3H, EE + AA, 0, 1);
        gemm(h_bf, HH, decWhh_bf, HH, dec_bhh, gh_buf, nullptr, G3H,
             BB, G3H, HH, 0, 1);
        k_gru<<<cdiv(BB * HH, 256), 256, 0, stream>>>(
            gi_dec, G3H, gh_buf, h_f32, h_bf, nullptr, 0);
        k_attention<<<BB, TT, 0, stream>>>(
            enc_out, h_f32, cat_bf, w_out + (size_t)s * BB * TT);
        gemm(cat_bf, 2 * HH, attnW_bf, 2 * HH, attn_b, nullptr, av_bf, AA,
             BB, AA, 2 * HH, 1 /*tanh*/, 1);
        gemm(av_bf, AA, clsW_bf, AA, cls_b,
             preds_out + (size_t)s * BB * VV, nullptr, VV,
             BB, VV, AA, 0, 4);
        k_argmax<<<BB, 256, 0, stream>>>(preds_out + (size_t)s * BB * VV, ids);
    }
}
